// Attention_6219112644965
// MI455X (gfx1250) — matrix operand order repaired and checked by reading, not compiled, not run on hardware
//
#include <hip/hip_runtime.h>
#include <hip/hip_bf16.h>

// ---------------------------------------------------------------------------
// MI455X / gfx1250 GQA attention layer, bf16 WMMA (16x16x32, f32 accum),
// Tensor Data Mover (TDM) double-buffered global->LDS staging.
// Pipeline: convert/transposes -> QKV GEMMs (V written transposed) ->
//           fused RMSNorm+RoPE -> flash attention -> output GEMM (f32 out).
// ---------------------------------------------------------------------------

typedef __attribute__((ext_vector_type(16))) __bf16 bf16x16;
typedef __attribute__((ext_vector_type(8)))  float  f32x8;
typedef __attribute__((ext_vector_type(4)))  unsigned int u32x4;
typedef __attribute__((ext_vector_type(8)))  int          i32x8;
typedef __attribute__((ext_vector_type(4)))  int          i32x4;

union FragAB { bf16x16 v; unsigned int u[8]; };

__device__ __forceinline__ unsigned short f32_to_bf16(float f) {
  unsigned int u = __float_as_uint(f);
  u += 0x7fffu + ((u >> 16) & 1u);          // round-to-nearest-even
  return (unsigned short)(u >> 16);
}
__device__ __forceinline__ float bf16_to_f32(unsigned int lo16) {
  return __uint_as_float(lo16 << 16);
}

__device__ __forceinline__ f32x8 wmma_bf16(const FragAB& a, const FragAB& b, f32x8 c) {
  return __builtin_amdgcn_wmma_f32_16x16x32_bf16(false, a.v, false, b.v,
                                                 (short)0, c, false, false);
}

// A-fragment (16x32 bf16): lane m = lane&15 holds row m; per-VGPR k-pair index
// = (j&3) + (j>>2)*8 + half*4   (ISA 7.12.2 16-bit A layout).
__device__ __forceinline__ void frag_load_a(FragAB& f, const unsigned short* row, int half) {
  const unsigned int* p = (const unsigned int*)row;   // row: k-contiguous bf16
#pragma unroll
  for (int j = 0; j < 8; ++j) f.u[j] = p[(j & 3) + ((j >> 2) << 3) + (half << 2)];
}
// B-fragment (32x16 bf16): lane n = lane&15 holds column n; k-pair index = (j&3) + (j>>2)*8 + half*4.
__device__ __forceinline__ void frag_load_b(FragAB& f, const unsigned short* row, int half) {
  const unsigned int* p = (const unsigned int*)row;   // row: k-contiguous bf16
#pragma unroll
  for (int j = 0; j < 8; ++j) f.u[j] = p[(j & 3) + ((j >> 2) << 3) + (half << 2)];
}

__device__ __forceinline__ f32x8 zero8() {
  f32x8 z;
#pragma unroll
  for (int i = 0; i < 8; ++i) z[i] = 0.0f;
  return z;
}

// ---------------------------------------------------------------------------
// TDM: issue a 2D bf16 tile load (tile_d0 contiguous elems x tile_d1 rows,
// row stride = stride_e elems) from global into LDS at lds_addr.
// D# encoding per CDNA5 ISA. 6-arg builtin
// (this toolchain): (g0, g1, g2, g3, g4, cpol) with zeroed groups for 2D.
// ---------------------------------------------------------------------------
__device__ __forceinline__ void tdm_load_2d(unsigned lds_addr, const void* gptr,
                                            unsigned tile_d0, unsigned tile_d1,
                                            unsigned long long stride_e,
                                            unsigned tensor_d0, unsigned tensor_d1) {
  const unsigned long long ga = (unsigned long long)gptr;
  u32x4 g0;
  g0.x = 1u;                                            // count=1, user descriptor
  g0.y = lds_addr;                                      // LDS byte address
  g0.z = (unsigned)ga;                                  // global_addr[31:0]
  g0.w = (unsigned)((ga >> 32) & 0x1FFFFFFu) | (2u << 30);  // addr[56:32] | type=2
  i32x8 g1;
  g1[0] = 0x10000;                                      // data_size=1 (2 bytes)
  g1[1] = (int)((tensor_d0 & 0xFFFFu) << 16);           // tensor_dim0[15:0]
  g1[2] = (int)(((tensor_d0 >> 16) & 0xFFFFu) | ((tensor_d1 & 0xFFFFu) << 16));
  g1[3] = (int)(((tensor_d1 >> 16) & 0xFFFFu) | (tile_d0 << 16));   // tile_dim0
  g1[4] = (int)tile_d1;                                 // tile_dim1 (tile_dim2=0)
  g1[5] = (int)(unsigned)(stride_e & 0xFFFFFFFFull);    // tensor_dim0_stride[31:0]
  g1[6] = (int)(unsigned)((stride_e >> 32) & 0xFFFFull);
  g1[7] = 0;
  const i32x4 gz4 = {0, 0, 0, 0};                       // 2D: groups 2/3 unused
  const i32x8 gz8 = {0, 0, 0, 0, 0, 0, 0, 0};
  __builtin_amdgcn_tensor_load_to_lds(g0, g1, gz4, gz4, gz8, 0);
}
__device__ __forceinline__ unsigned lds_addr_of(const void* p) {
  return (unsigned)(unsigned long long)p;               // low 32 bits = LDS offset
}

// ---------------------------------------------------------------------------
// f32 -> bf16 bulk convert (vectorized, grid-stride)
// ---------------------------------------------------------------------------
__global__ void f32_to_bf16_kernel(const float* __restrict__ in,
                                   unsigned short* __restrict__ out, long long n4) {
  long long i = (long long)blockIdx.x * blockDim.x + threadIdx.x;
  const long long stride = (long long)gridDim.x * blockDim.x;
  for (; i < n4; i += stride) {
    const float4 f = ((const float4*)in)[i];
    union { unsigned short s[4]; uint2 u; } r;
    r.s[0] = f32_to_bf16(f.x); r.s[1] = f32_to_bf16(f.y);
    r.s[2] = f32_to_bf16(f.z); r.s[3] = f32_to_bf16(f.w);
    ((uint2*)out)[i] = r.u;
  }
}

// ---------------------------------------------------------------------------
// Tiled transpose + convert: in[K][N] f32 -> out[N][K] bf16 (one-time, weights)
// ---------------------------------------------------------------------------
__global__ __launch_bounds__(256) void transpose_f32_to_bf16(
    const float* __restrict__ in, unsigned short* __restrict__ out, int K, int N) {
  __shared__ unsigned short tile[32][33];
  const int k0 = blockIdx.y * 32;
  const int n0 = blockIdx.x * 32;
  const int tx = threadIdx.x & 31;
  const int ty = threadIdx.x >> 5;      // 0..7
#pragma unroll
  for (int i = 0; i < 4; ++i) {
    const int k = ty + i * 8;
    tile[tx][k] = f32_to_bf16(in[(size_t)(k0 + k) * N + n0 + tx]);
  }
  __syncthreads();
#pragma unroll
  for (int i = 0; i < 4; ++i) {
    const int n = ty + i * 8;
    out[(size_t)(n0 + n) * K + k0 + tx] = tile[n][tx];
  }
}

// ---------------------------------------------------------------------------
// bf16 GEMM: C[M,N] = A[M,K] * Bt[N,K]^T;  BM=BN=128, BK=32, TDM double-buffer.
// 256 threads = 8 waves (4 row-waves x 2 col-waves), each wave 32x64.
// out_mode: 0 = f32 row-major; 1 = bf16 row-major;
//           2 = bf16 "V-transposed": out[(b*N+col)*Tp + t], b=row/Tp, t=row%Tp
// ---------------------------------------------------------------------------
__global__ __launch_bounds__(256) void gemm_bf16(
    const unsigned short* __restrict__ A, const unsigned short* __restrict__ Bt,
    void* __restrict__ Cout, int M, int N, int K, int out_mode, int Tp) {
  __shared__ unsigned short As[2][128 * 32];   // [m][k]
  __shared__ unsigned short Bs[2][128 * 32];   // [n][k]

  const int tid  = threadIdx.x;
  const int lane = tid & 31;
  const int wave = tid >> 5;
  const int half = lane >> 4;
  const int ln   = lane & 15;
  const int wr   = wave >> 1;     // 0..3
  const int wc   = wave & 1;      // 0..1
  const int blockRow = blockIdx.y * 128;
  const int blockCol = blockIdx.x * 128;

  f32x8 acc[2][4];
#pragma unroll
  for (int mt = 0; mt < 2; ++mt)
#pragma unroll
    for (int nt = 0; nt < 4; ++nt) acc[mt][nt] = zero8();

  const int kTiles = K >> 5;
  auto issue = [&](int kt, int buf) {
    if (wave == 0) {
      tdm_load_2d(lds_addr_of(&As[buf][0]), A  + (size_t)blockRow * K + (kt << 5),
                  32, 128, (unsigned)K, (unsigned)K, 0x100000u);
      tdm_load_2d(lds_addr_of(&Bs[buf][0]), Bt + (size_t)blockCol * K + (kt << 5),
                  32, 128, (unsigned)K, (unsigned)K, 0x100000u);
    }
  };

  issue(0, 0);
  if (wave == 0) __builtin_amdgcn_s_wait_tensorcnt(0);
  __syncthreads();

  for (int kt = 0; kt < kTiles; ++kt) {
    const int buf = kt & 1;
    if (kt + 1 < kTiles) issue(kt + 1, buf ^ 1);   // DMA next tile during WMMAs

    FragAB a[2], bfr[4];
#pragma unroll
    for (int mt = 0; mt < 2; ++mt)
      frag_load_a(a[mt], &As[buf][(wr * 32 + mt * 16 + ln) * 32], half);
#pragma unroll
    for (int nt = 0; nt < 4; ++nt)
      frag_load_b(bfr[nt], &Bs[buf][(wc * 64 + nt * 16 + ln) * 32], half);
#pragma unroll
    for (int mt = 0; mt < 2; ++mt)
#pragma unroll
      for (int nt = 0; nt < 4; ++nt)
        acc[mt][nt] = wmma_bf16(a[mt], bfr[nt], acc[mt][nt]);

    if (kt + 1 < kTiles && wave == 0) __builtin_amdgcn_s_wait_tensorcnt(0);
    __syncthreads();
  }

#pragma unroll
  for (int mt = 0; mt < 2; ++mt)
#pragma unroll
    for (int nt = 0; nt < 4; ++nt) {
      const int row = blockRow + wr * 32 + mt * 16 + half * 8;  // + r
      const int col = blockCol + wc * 64 + nt * 16 + ln;
      if (out_mode == 2) {          // packed b128 store of 8 consecutive t
        const int b  = row / Tp;
        const int t0 = row % Tp;
        union { unsigned short s[8]; uint4 u; } pk;
#pragma unroll
        for (int r = 0; r < 8; ++r) pk.s[r] = f32_to_bf16(acc[mt][nt][r]);
        *(uint4*)((unsigned short*)Cout + ((size_t)b * N + col) * Tp + t0) = pk.u;
      } else if (out_mode == 1) {
#pragma unroll
        for (int r = 0; r < 8; ++r)
          ((unsigned short*)Cout)[(size_t)(row + r) * N + col] = f32_to_bf16(acc[mt][nt][r]);
      } else {
#pragma unroll
        for (int r = 0; r < 8; ++r)
          ((float*)Cout)[(size_t)(row + r) * N + col] = acc[mt][nt][r];
      }
    }
}

// ---------------------------------------------------------------------------
// Fused RMSNorm + interleaved RoPE, in-place on bf16 rows of HD=128.
// One wave per (b,t,head) row; outscale folds 1/sqrt(HD) into Q.
// ---------------------------------------------------------------------------
__global__ __launch_bounds__(256) void rmsnorm_rope(
    unsigned short* __restrict__ X, const float* __restrict__ cosT,
    const float* __restrict__ sinT, const float* __restrict__ gamma,
    int T, int NH, float outscale) {
  const int lane = threadIdx.x & 31;
  const int wave = threadIdx.x >> 5;
  const long long row = (long long)blockIdx.x * 8 + wave;
  const int t = (int)((row / NH) % T);
  unsigned short* p = X + row * 128 + lane * 4;

  uint2 d = *(const uint2*)p;
  float v[4];
  v[0] = bf16_to_f32(d.x & 0xffffu); v[1] = bf16_to_f32(d.x >> 16);
  v[2] = bf16_to_f32(d.y & 0xffffu); v[3] = bf16_to_f32(d.y >> 16);

  float ss = v[0] * v[0] + v[1] * v[1] + v[2] * v[2] + v[3] * v[3];
#pragma unroll
  for (int m = 16; m >= 1; m >>= 1) ss += __shfl_xor(ss, m, 32);
  const float rinv = rsqrtf(ss * (1.0f / 128.0f) + 1e-6f);

  union { unsigned short s[4]; uint2 u; } out;
#pragma unroll
  for (int pr = 0; pr < 2; ++pr) {
    const int i = lane * 4 + 2 * pr;
    const float c = cosT[t * 128 + i];
    const float s = sinT[t * 128 + i];
    const float h0 = v[2 * pr]     * rinv * gamma[i];
    const float h1 = v[2 * pr + 1] * rinv * gamma[i + 1];
    out.s[2 * pr]     = f32_to_bf16((h0 * c - h1 * s) * outscale);
    out.s[2 * pr + 1] = f32_to_bf16((h1 * c + h0 * s) * outscale);
  }
  *(uint2*)p = out.u;
}

// ---------------------------------------------------------------------------
// Flash attention (non-causal, online softmax). Grid: (T/128, H, B).
// 8 waves; wave owns 16 Q rows (Q in registers). Bc = 64 keys per block.
// K staged [kc][d] via TDM; V pre-transposed in global (v_t[(b,kv,d)][t]) so
// its tile [d][kc] is also a plain TDM 2D tile. Double-buffered.
// ---------------------------------------------------------------------------
__global__ __launch_bounds__(256) void flash_attn(
    const unsigned short* __restrict__ Q, const unsigned short* __restrict__ Kb,
    const unsigned short* __restrict__ Vt, unsigned short* __restrict__ O,
    int Bn, int T, int Hh, int KVh) {
  __shared__ unsigned short Ks[2][64 * 128];   // [kc][d]
  __shared__ unsigned short Vs[2][128 * 64];   // [d][kc]
  __shared__ unsigned short Ps[8][16 * 64];    // per-wave P patch [m][kc]

  const int tid  = threadIdx.x;
  const int lane = tid & 31;
  const int wave = tid >> 5;
  const int half = lane >> 4;
  const int ln   = lane & 15;
  const int b    = blockIdx.z;
  const int h    = blockIdx.y;
  const int kvh  = h / (Hh / KVh);
  const int qrow0 = blockIdx.x * 128 + wave * 16;

  // Q fragments for this wave's 16 rows (softmax scale already folded in).
  FragAB qf[4];
  {
    const size_t qbase = (((size_t)b * T + qrow0 + ln) * Hh + h) * 128;
#pragma unroll
    for (int f = 0; f < 4; ++f) {
      const unsigned int* p = (const unsigned int*)(Q + qbase + f * 32);
#pragma unroll
      for (int j = 0; j < 8; ++j)
        qf[f].u[j] = p[(j & 3) + ((j >> 2) << 3) + (half << 2)];
    }
  }

  f32x8 oacc[8];
#pragma unroll
  for (int dt = 0; dt < 8; ++dt) oacc[dt] = zero8();
  float mrow[8], lrow[8];
#pragma unroll
  for (int r = 0; r < 8; ++r) { mrow[r] = -3.0e38f; lrow[r] = 0.0f; }

  const int nKb = T / 64;
  auto issueKV = [&](int kb, int buf) {
    if (wave == 0) {
      tdm_load_2d(lds_addr_of(&Ks[buf][0]),
                  Kb + (((size_t)b * T + kb * 64) * KVh + kvh) * 128,
                  128, 64, (unsigned)(KVh * 128), (unsigned)(KVh * 128), 0x100000u);
      tdm_load_2d(lds_addr_of(&Vs[buf][0]),
                  Vt + ((size_t)(b * KVh + kvh) * 128) * T + kb * 64,
                  64, 128, (unsigned)T, (unsigned)T, 0x100000u);
    }
  };

  issueKV(0, 0);
  if (wave == 0) __builtin_amdgcn_s_wait_tensorcnt(0);
  __syncthreads();

  for (int kb = 0; kb < nKb; ++kb) {
    const int buf = kb & 1;
    if (kb + 1 < nKb) issueKV(kb + 1, buf ^ 1);   // DMA next K/V during compute

    // S = Q * K^T : 4 n-tiles x 4 k-steps = 16 WMMA
    f32x8 sacc[4];
#pragma unroll
    for (int nt = 0; nt < 4; ++nt) {
      f32x8 c = zero8();
#pragma unroll
      for (int ks = 0; ks < 4; ++ks) {
        FragAB bf;
        frag_load_b(bf, &Ks[buf][(nt * 16 + ln) * 128 + ks * 32], half);
        c = wmma_bf16(qf[ks], bf, c);
      }
      sacc[nt] = c;
    }

    // online softmax: row r lives at m_local = r + half*8; reductions width 16
    float alpha[8];
#pragma unroll
    for (int r = 0; r < 8; ++r) {
      float mx = sacc[0][r];
#pragma unroll
      for (int nt = 1; nt < 4; ++nt) mx = fmaxf(mx, sacc[nt][r]);
#pragma unroll
      for (int m = 8; m >= 1; m >>= 1) mx = fmaxf(mx, __shfl_xor(mx, m, 16));
      const float mnew = fmaxf(mrow[r], mx);
      alpha[r] = __expf(mrow[r] - mnew);
      mrow[r] = mnew;
    }
#pragma unroll
    for (int nt = 0; nt < 4; ++nt)
#pragma unroll
      for (int r = 0; r < 8; ++r) {
        const float pv = __expf(sacc[nt][r] - mrow[r]);
        sacc[nt][r] = pv;
        Ps[wave][(half * 8 + r) * 64 + nt * 16 + ln] = f32_to_bf16(pv);
      }
#pragma unroll
    for (int r = 0; r < 8; ++r) {
      float s = sacc[0][r] + sacc[1][r] + sacc[2][r] + sacc[3][r];
#pragma unroll
      for (int m = 8; m >= 1; m >>= 1) s += __shfl_xor(s, m, 16);
      lrow[r] = lrow[r] * alpha[r] + s;
#pragma unroll
      for (int dt = 0; dt < 8; ++dt) oacc[dt][r] *= alpha[r];
    }

    // O += P * V : P re-packed to A-layout via wave-private LDS patch
    FragAB pf[2];
#pragma unroll
    for (int ks = 0; ks < 2; ++ks)
      frag_load_a(pf[ks], &Ps[wave][ln * 64 + ks * 32], half);
#pragma unroll
    for (int dt = 0; dt < 8; ++dt)
#pragma unroll
      for (int ks = 0; ks < 2; ++ks) {
        FragAB vf;
        frag_load_b(vf, &Vs[buf][(dt * 16 + ln) * 64 + ks * 32], half);
        oacc[dt] = wmma_bf16(pf[ks], vf, oacc[dt]);
      }

    if (kb + 1 < nKb && wave == 0) __builtin_amdgcn_s_wait_tensorcnt(0);
    __syncthreads();
  }

  // finalize: O /= l, write bf16 [b,t,h,d]
#pragma unroll
  for (int r = 0; r < 8; ++r) {
    const float inv = 1.0f / lrow[r];
    const int trow = qrow0 + half * 8 + r;
#pragma unroll
    for (int dt = 0; dt < 8; ++dt)
      O[(((size_t)b * T + trow) * Hh + h) * 128 + dt * 16 + ln] =
          f32_to_bf16(oacc[dt][r] * inv);
  }
}

// ---------------------------------------------------------------------------
// Host-side orchestration
// ---------------------------------------------------------------------------
extern "C" void kernel_launch(void* const* d_in, const int* in_sizes, int n_in,
                              void* d_out, int out_size, void* d_ws, size_t ws_size,
                              hipStream_t stream) {
  (void)in_sizes; (void)n_in; (void)out_size; (void)ws_size;
  const float* x    = (const float*)d_in[0];
  const float* cosT = (const float*)d_in[1];
  const float* sinT = (const float*)d_in[2];
  const float* Wq   = (const float*)d_in[3];
  const float* Wk   = (const float*)d_in[4];
  const float* Wv   = (const float*)d_in[5];
  const float* Wo   = (const float*)d_in[6];
  const float* qg   = (const float*)d_in[7];
  const float* kg   = (const float*)d_in[8];

  const int Bn = 4, T = 2048, D = 2048, H = 16, KV = 8;
  const long long M = (long long)Bn * T;   // 8192 rows

  char* w = (char*)d_ws;
  unsigned short* x_bf  = (unsigned short*)w; w += (size_t)M * D * 2;            // 33.5 MB
  unsigned short* WqT   = (unsigned short*)w; w += (size_t)D * (H * 128) * 2;    //  8.4 MB  [N][K]
  unsigned short* WkT   = (unsigned short*)w; w += (size_t)D * (KV * 128) * 2;   //  4.2 MB
  unsigned short* WvT   = (unsigned short*)w; w += (size_t)D * (KV * 128) * 2;   //  4.2 MB
  unsigned short* WoT   = (unsigned short*)w; w += (size_t)(H * 128) * D * 2;    //  8.4 MB
  unsigned short* q_bf  = (unsigned short*)w; w += (size_t)M * H * 128 * 2;      // 33.5 MB
  unsigned short* k_bf  = (unsigned short*)w; w += (size_t)M * KV * 128 * 2;     // 16.8 MB
  unsigned short* v_t   = (unsigned short*)w; w += (size_t)M * KV * 128 * 2;     // 16.8 MB [b,kv,d][t]
  unsigned short* o_bf  = (unsigned short*)w; w += (size_t)M * H * 128 * 2;      // 33.5 MB

  // 1) converts; weights converted *transposed* to [N][K] for k-contiguous staging
  f32_to_bf16_kernel<<<4096, 256, 0, stream>>>(x, x_bf, M * D / 4);
  transpose_f32_to_bf16<<<dim3(H * 128 / 32,  D / 32), 256, 0, stream>>>(Wq, WqT, D, H * 128);
  transpose_f32_to_bf16<<<dim3(KV * 128 / 32, D / 32), 256, 0, stream>>>(Wk, WkT, D, KV * 128);
  transpose_f32_to_bf16<<<dim3(KV * 128 / 32, D / 32), 256, 0, stream>>>(Wv, WvT, D, KV * 128);
  transpose_f32_to_bf16<<<dim3(D / 32, H * 128 / 32), 256, 0, stream>>>(Wo, WoT, H * 128, D);

  // 2) QKV projections (V epilogue writes transposed v_t directly)
  gemm_bf16<<<dim3(2048 / 128, M / 128), 256, 0, stream>>>(x_bf, WqT, q_bf, (int)M, 2048, 2048, 1, T);
  gemm_bf16<<<dim3(1024 / 128, M / 128), 256, 0, stream>>>(x_bf, WkT, k_bf, (int)M, 1024, 2048, 1, T);
  gemm_bf16<<<dim3(1024 / 128, M / 128), 256, 0, stream>>>(x_bf, WvT, v_t,  (int)M, 1024, 2048, 2, T);

  // 3) RMSNorm + RoPE in place (1/sqrt(HD) folded into Q)
  rmsnorm_rope<<<(unsigned)(M * H  / 8), 256, 0, stream>>>(q_bf, cosT, sinT, qg, T, H,  0.08838834764831843f);
  rmsnorm_rope<<<(unsigned)(M * KV / 8), 256, 0, stream>>>(k_bf, cosT, sinT, kg, T, KV, 1.0f);

  // 4) attention
  flash_attn<<<dim3(T / 128, H, Bn), 256, 0, stream>>>(q_bf, k_bf, v_t, o_bf, Bn, T, H, KV);

  // 5) output projection, fp32 straight to d_out
  gemm_bf16<<<dim3(2048 / 128, M / 128), 256, 0, stream>>>(o_bf, WoT, d_out, (int)M, 2048, 2048, 0, T);
}
